// HGNN_59828894433623
// MI455X (gfx1250) — matrix operand order repaired and checked by reading, not compiled, not run on hardware
//
#include <hip/hip_runtime.h>
#include <hip/hip_bf16.h>

// ---------------------------------------------------------------------------
// CDNA5 (gfx1250) hetero-SAGE pipeline.
//  - bf16 WMMA GEMMs (v_wmma_f32_16x16x32_bf16), fp32 accumulate
//  - weights pre-swizzled once into WMMA B-lane layout (bf16, 32B/lane reads)
//  - fully templated GEMM variants (no runtime flag code), guard-free fast path
//  - compile-time leading dimensions -> 32-bit shift addressing
//  - paper update fused into one 3-operand GEMM via summed right-weights
//  - wave-per-edge coalesced gather + native fp32 atomic scatter
// ---------------------------------------------------------------------------

typedef __attribute__((ext_vector_type(16))) __bf16 v16bf;
typedef __attribute__((ext_vector_type(8)))  float  v8f;

#define HDIM 64

// Pack two floats into two bf16 (low = x, high = y) using HW converts.
static __device__ __forceinline__ unsigned pk2(float x, float y) {
  __hip_bfloat162 h = __float22bfloat162_rn(make_float2(x, y));
  unsigned r;
  __builtin_memcpy(&r, &h, sizeof(r));
  return r;
}

// A-matrix 16x32 bf16 tile (ISA 7.12.2): lane m=l&15 holds row m; half hi=l>>4
// holds K runs {hi*8..hi*8+7} and {16+hi*8..16+hi*8+7} -> two contiguous
// 32-byte runs per lane, loaded as float4s. LD is the compile-time leading dim.
template <bool GUARD, bool CNT, int LD>
static __device__ __forceinline__
v16bf load_a_tile(const float* __restrict__ A, int rowbase, int kbase,
                  int Nrows, int m, int hi, const float* __restrict__ cntp) {
  v16bf r;
  const int row = rowbase + m;
  if (!GUARD || row < Nrows) {
    float s = 1.0f;
    if (CNT) s = 1.0f / fmaxf(cntp[row], 1.0f);
    const float4* q =
        (const float4*)(A + (size_t)((unsigned)row * (unsigned)LD) + kbase + hi * 8);
    const float4 f0 = q[0], f1 = q[1], f2 = q[4], f3 = q[5];
    r[0]  = (__bf16)(f0.x * s);
    r[1]  = (__bf16)(f0.y * s);
    r[2]  = (__bf16)(f0.z * s);
    r[3]  = (__bf16)(f0.w * s);
    r[4]  = (__bf16)(f1.x * s);
    r[5]  = (__bf16)(f1.y * s);
    r[6]  = (__bf16)(f1.z * s);
    r[7]  = (__bf16)(f1.w * s);
    r[8]  = (__bf16)(f2.x * s);
    r[9]  = (__bf16)(f2.y * s);
    r[10] = (__bf16)(f2.z * s);
    r[11] = (__bf16)(f2.w * s);
    r[12] = (__bf16)(f3.x * s);
    r[13] = (__bf16)(f3.y * s);
    r[14] = (__bf16)(f3.z * s);
    r[15] = (__bf16)(f3.w * s);
  } else {
#pragma unroll
    for (int e = 0; e < 16; ++e) r[e] = (__bf16)0.0f;
  }
  return r;
}

// out[N,HOUT] = act( sum_i (Ai (/cnti)) @ Wi + bias (+ bias2) ), up to 3 operands.
// Operand i has K = 32*KSi and leading dimension 32*KSi.
template <int HOUT, int KS1, bool CNT1, int KS2, bool CNT2, int KS3,
          bool RELU, bool BIAS2, bool GUARD>
static __device__ __forceinline__
void gemm_body(const float* __restrict__ A1,
               const float* __restrict__ cnt1, const v16bf* __restrict__ W1,
               const float* __restrict__ A2,
               const float* __restrict__ cnt2, const v16bf* __restrict__ W2,
               const float* __restrict__ A3, const v16bf* __restrict__ W3,
               const float* __restrict__ bias, const float* __restrict__ bias2,
               float* __restrict__ out, int Nrows, int rowbase,
               int lane, int m, int hi) {
  constexpr int NT = HOUT / 16;

  v16bf a1[2], a2[2], a3[2];
#pragma unroll
  for (int s = 0; s < KS1; ++s)
    a1[s] = load_a_tile<GUARD, CNT1, 32 * KS1>(A1, rowbase, 32 * s, Nrows, m, hi, cnt1);
  if constexpr (KS2 > 0) {
#pragma unroll
    for (int s = 0; s < KS2; ++s)
      a2[s] = load_a_tile<GUARD, CNT2, 32 * KS2>(A2, rowbase, 32 * s, Nrows, m, hi, cnt2);
  }
  if constexpr (KS3 > 0) {
#pragma unroll
    for (int s = 0; s < KS3; ++s)
      a3[s] = load_a_tile<GUARD, false, 32 * KS3>(A3, rowbase, 32 * s, Nrows, m, hi, nullptr);
  }

#pragma unroll
  for (int c = 0; c < NT; ++c) {
    const int nbase = c * 16;
    float bb = bias[nbase + m];
    if constexpr (BIAS2) bb += bias2[nbase + m];
    v8f acc;
#pragma unroll
    for (int r = 0; r < 8; ++r) acc[r] = bb;

#pragma unroll
    for (int s = 0; s < KS1; ++s) {
      v16bf b = W1[(s * NT + c) * 32 + lane];
      acc = __builtin_amdgcn_wmma_f32_16x16x32_bf16(false, a1[s], false, b,
                                                    (short)0, acc, false, false);
    }
    if constexpr (KS2 > 0) {
#pragma unroll
      for (int s = 0; s < KS2; ++s) {
        v16bf b = W2[(s * NT + c) * 32 + lane];
        acc = __builtin_amdgcn_wmma_f32_16x16x32_bf16(false, a2[s], false, b,
                                                      (short)0, acc, false, false);
      }
    }
    if constexpr (KS3 > 0) {
#pragma unroll
      for (int s = 0; s < KS3; ++s) {
        v16bf b = W3[(s * NT + c) * 32 + lane];
        acc = __builtin_amdgcn_wmma_f32_16x16x32_bf16(false, a3[s], false, b,
                                                      (short)0, acc, false, false);
      }
    }

#pragma unroll
    for (int r = 0; r < 8; ++r) {
      const int row = rowbase + r + 8 * hi;
      if (!GUARD || row < Nrows) {
        float v = acc[r];
        if constexpr (RELU) v = fmaxf(v, 0.0f);
        out[(size_t)((unsigned)row * (unsigned)HOUT) + nbase + m] = v;
      }
    }
  }
}

// 8 waves per block, 16 rows per wave.
template <int HOUT, int KS1, bool CNT1, int KS2, bool CNT2, int KS3,
          bool RELU, bool BIAS2>
__global__ __launch_bounds__(256)
void hgnn_gemm_kernel(const float* __restrict__ A1,
                      const float* __restrict__ cnt1, const v16bf* __restrict__ W1,
                      const float* __restrict__ A2,
                      const float* __restrict__ cnt2, const v16bf* __restrict__ W2,
                      const float* __restrict__ A3, const v16bf* __restrict__ W3,
                      const float* __restrict__ bias, const float* __restrict__ bias2,
                      float* __restrict__ out, int Nrows) {
  const int lane = threadIdx.x & 31;
  const int wave = threadIdx.x >> 5;
  const int m    = lane & 15;
  const int hi   = lane >> 4;
  const int rowbase = (blockIdx.x * 8 + wave) * 16;
  if (rowbase >= Nrows) return;     // wave-uniform: EXEC all-1s for WMMA
  if (rowbase + 16 <= Nrows) {      // uniform: full tile, no per-lane guards
    gemm_body<HOUT, KS1, CNT1, KS2, CNT2, KS3, RELU, BIAS2, false>(
        A1, cnt1, W1, A2, cnt2, W2, A3, W3, bias, bias2, out, Nrows,
        rowbase, lane, m, hi);
  } else {
    gemm_body<HOUT, KS1, CNT1, KS2, CNT2, KS3, RELU, BIAS2, true>(
        A1, cnt1, W1, A2, cnt2, W2, A3, W3, bias, bias2, out, Nrows,
        rowbase, lane, m, hi);
  }
}

// Pre-swizzle a K x N fp32 weight matrix (optionally summed with Wadd) into
// bf16 WMMA B-operand layout: tile = ks*(N/16)+c; lane l (n=l&15, hi=l>>4)
// holds 16 bf16, element e -> W[ks*32 + hi*8 + e + (e>=8 ? 8 : 0)][c*16 + n]. One wave/tile.
__global__ __launch_bounds__(32)
void hgnn_swizzle_w_kernel(const float* __restrict__ W,
                           const float* __restrict__ Wadd, int N,
                           unsigned* __restrict__ out) {
  const int ct = N >> 4;
  const int tile = blockIdx.x;
  const int ks = tile / ct, c = tile % ct;
  const int lane = threadIdx.x & 31;
  const int n = lane & 15, hi = lane >> 4;
  const size_t base = (size_t)(ks * 32 + hi * 8) * N + c * 16 + n;
  unsigned* o = out + (size_t)tile * 256 + lane * 8;
#pragma unroll
  for (int i = 0; i < 8; ++i) {
    const int r0 = 2 * i + (i >= 4 ? 8 : 0);
    float x0 = W[base + (size_t)(r0) * N];
    float x1 = W[base + (size_t)(r0 + 1) * N];
    if (Wadd) {
      x0 += Wadd[base + (size_t)(r0) * N];
      x1 += Wadd[base + (size_t)(r0 + 1) * N];
    }
    o[i] = pk2(x0, x1);
  }
}

// One wave per edge: lane i moves elements 2i, 2i+1 of the 64-float feature.
__global__ __launch_bounds__(256)
void hgnn_edge_agg_kernel(const float* __restrict__ ysrc,
                          const int* __restrict__ esrc,
                          const int* __restrict__ edst,
                          float* __restrict__ agg,
                          float* __restrict__ cnt, int E) {
  const int lane = threadIdx.x & 31;
  const int e = (int)(blockIdx.x * (blockDim.x >> 5) + (threadIdx.x >> 5));
  if (e >= E) return;
  const int s = esrc[e];
  const int d = edst[e];
  const float2 v = *(const float2*)(ysrc + (size_t)((unsigned)s * (unsigned)HDIM) + lane * 2);
  float* dst = agg + (size_t)((unsigned)d * (unsigned)HDIM) + lane * 2;
  __hip_atomic_fetch_add(dst + 0, v.x, __ATOMIC_RELAXED, __HIP_MEMORY_SCOPE_AGENT);
  __hip_atomic_fetch_add(dst + 1, v.y, __ATOMIC_RELAXED, __HIP_MEMORY_SCOPE_AGENT);
  if (lane == 0)
    __hip_atomic_fetch_add(cnt + d, 1.0f, __ATOMIC_RELAXED, __HIP_MEMORY_SCOPE_AGENT);
}

extern "C" void kernel_launch(void* const* d_in, const int* in_sizes, int n_in,
                              void* d_out, int out_size, void* d_ws, size_t ws_size,
                              hipStream_t stream) {
  (void)n_in; (void)out_size; (void)ws_size;
  const float* x_d_a    = (const float*)d_in[0];
  const float* x_c_a    = (const float*)d_in[1];
  const float* x_d_p    = (const float*)d_in[2];
  const float* x_c_p    = (const float*)d_in[3];
  const float* win_a_w  = (const float*)d_in[4];
  const float* win_a_b  = (const float*)d_in[5];
  const float* win_p_w  = (const float*)d_in[6];
  const float* win_p_b  = (const float*)d_in[7];
  const float* writes_wl = (const float*)d_in[8];
  const float* writes_bl = (const float*)d_in[9];
  const float* writes_wr = (const float*)d_in[10];
  const float* wb_wl     = (const float*)d_in[11];
  const float* wb_bl     = (const float*)d_in[12];
  const float* wb_wr     = (const float*)d_in[13];
  const float* cites_wl  = (const float*)d_in[14];
  const float* cites_bl  = (const float*)d_in[15];
  const float* cites_wr  = (const float*)d_in[16];
  const float* lin_w     = (const float*)d_in[17];
  const float* lin_b     = (const float*)d_in[18];
  const int* writes_src  = (const int*)d_in[19];
  const int* writes_dst  = (const int*)d_in[20];
  const int* wb_src      = (const int*)d_in[21];
  const int* wb_dst      = (const int*)d_in[22];
  const int* cites_src   = (const int*)d_in[23];
  const int* cites_dst   = (const int*)d_in[24];

  const int NA = in_sizes[0] / 32;
  const int NP = in_sizes[2] / 32;
  const int E  = in_sizes[19];
  const int H = HDIM, L = 2;
  const size_t maxN = (size_t)(NA > NP ? NA : NP);

  float* wsf = (float*)d_ws;
  float* y_a   = wsf;  wsf += (size_t)NA * H;
  float* y_p   = wsf;  wsf += (size_t)NP * H;
  float* n_a   = wsf;  wsf += (size_t)NA * H;
  float* n_p   = wsf;  wsf += (size_t)NP * H;
  float* agg_w = wsf;  wsf += maxN * H;
  float* cnt_w = wsf;  wsf += maxN;
  float* agg_c = wsf;  wsf += maxN * H;
  float* cnt_c = wsf;  wsf += maxN;

  // Swizzled bf16 weight region (32B aligned).
  uintptr_t pa = ((uintptr_t)wsf + 31) & ~(uintptr_t)31;
  unsigned* sw = (unsigned*)pa;
  const size_t M64 = 2048;  // u32 words per swizzled 64x64 matrix (8 tiles)
  unsigned* sw_win_a = sw; sw += M64;
  unsigned* sw_win_p = sw; sw += M64;
  unsigned *sw_wwl[2], *sw_cwl[2], *sw_sumr[2], *sw_bwl[2], *sw_bwr[2];
  for (int l = 0; l < L; ++l) {
    sw_wwl[l]  = sw; sw += M64;
    sw_cwl[l]  = sw; sw += M64;
    sw_sumr[l] = sw; sw += M64;  // writes_wr + cites_wr
    sw_bwl[l]  = sw; sw += M64;
    sw_bwr[l]  = sw; sw += M64;
  }
  unsigned* sw_lin = sw; sw += 512;  // 64x16 -> 2 tiles

  const dim3 blk(256);
  const dim3 gA((unsigned)((NA + 127) / 128));
  const dim3 gP((unsigned)((NP + 127) / 128));
  const dim3 gE((unsigned)((E + 7) / 8));

  // --- swizzle all weights once (tiny) ---
  hgnn_swizzle_w_kernel<<<8, 32, 0, stream>>>(win_a_w, nullptr, 64, sw_win_a);
  hgnn_swizzle_w_kernel<<<8, 32, 0, stream>>>(win_p_w, nullptr, 64, sw_win_p);
  for (int l = 0; l < L; ++l) {
    const size_t o = (size_t)l * H * H;
    hgnn_swizzle_w_kernel<<<8, 32, 0, stream>>>(writes_wl + o, nullptr, 64, sw_wwl[l]);
    hgnn_swizzle_w_kernel<<<8, 32, 0, stream>>>(cites_wl + o, nullptr, 64, sw_cwl[l]);
    hgnn_swizzle_w_kernel<<<8, 32, 0, stream>>>(writes_wr + o, cites_wr + o, 64, sw_sumr[l]);
    hgnn_swizzle_w_kernel<<<8, 32, 0, stream>>>(wb_wl + o, nullptr, 64, sw_bwl[l]);
    hgnn_swizzle_w_kernel<<<8, 32, 0, stream>>>(wb_wr + o, nullptr, 64, sw_bwr[l]);
  }
  hgnn_swizzle_w_kernel<<<2, 32, 0, stream>>>(lin_w, nullptr, 16, sw_lin);

  // --- input projections: relu([x_d|x_c] @ W + b), concat split across operands
  hgnn_gemm_kernel<64, 1, false, 1, false, 0, true, false><<<gA, blk, 0, stream>>>(
      x_d_a, nullptr, (const v16bf*)sw_win_a,
      x_c_a, nullptr, (const v16bf*)(sw_win_a + 4 * 256),
      nullptr, nullptr, win_a_b, nullptr, y_a, NA);
  hgnn_gemm_kernel<64, 1, false, 1, false, 0, true, false><<<gP, blk, 0, stream>>>(
      x_d_p, nullptr, (const v16bf*)sw_win_p,
      x_c_p, nullptr, (const v16bf*)(sw_win_p + 4 * 256),
      nullptr, nullptr, win_p_b, nullptr, y_p, NP);

  float* ca = y_a; float* cp = y_p; float* na = n_a; float* npp = n_p;
  for (int l = 0; l < L; ++l) {
    const float* wbl = writes_bl + (size_t)l * H;
    const float* bbl = wb_bl    + (size_t)l * H;
    const float* cbl = cites_bl + (size_t)l * H;

    // aggregate writes (author->paper) and cites (paper->paper)
    hipMemsetAsync(agg_w, 0, maxN * H * sizeof(float), stream);
    hipMemsetAsync(cnt_w, 0, maxN * sizeof(float), stream);
    hipMemsetAsync(agg_c, 0, maxN * H * sizeof(float), stream);
    hipMemsetAsync(cnt_c, 0, maxN * sizeof(float), stream);
    hgnn_edge_agg_kernel<<<gE, blk, 0, stream>>>(ca, writes_src, writes_dst, agg_w, cnt_w, E);
    hgnn_edge_agg_kernel<<<gE, blk, 0, stream>>>(cp, cites_src, cites_dst, agg_c, cnt_c, E);

    // new_p = mean_w@wwl + mean_c@cwl + y_p@(wwr+cwr) + (wbl+cbl)
    hgnn_gemm_kernel<64, 2, true, 2, true, 2, false, true><<<gP, blk, 0, stream>>>(
        agg_w, cnt_w, (const v16bf*)sw_wwl[l],
        agg_c, cnt_c, (const v16bf*)sw_cwl[l],
        cp, (const v16bf*)sw_sumr[l],
        wbl, cbl, npp, NP);

    // aggregate written_by (paper->author), reuse agg_w/cnt_w
    hipMemsetAsync(agg_w, 0, maxN * H * sizeof(float), stream);
    hipMemsetAsync(cnt_w, 0, maxN * sizeof(float), stream);
    hgnn_edge_agg_kernel<<<gE, blk, 0, stream>>>(cp, wb_src, wb_dst, agg_w, cnt_w, E);

    // new_a = mean_wb@bwl + y_a@bwr + bbl
    hgnn_gemm_kernel<64, 2, true, 2, false, 0, false, false><<<gA, blk, 0, stream>>>(
        agg_w, cnt_w, (const v16bf*)sw_bwl[l],
        ca, nullptr, (const v16bf*)sw_bwr[l],
        nullptr, nullptr, bbl, nullptr, na, NA);

    float* t;
    t = ca; ca = na;  na  = t;
    t = cp; cp = npp; npp = t;
  }

  // Final linear: y_p @ lin_w + lin_b -> [NP, 16]
  hgnn_gemm_kernel<16, 2, false, 0, false, 0, false, false><<<gP, blk, 0, stream>>>(
      cp, nullptr, (const v16bf*)sw_lin,
      nullptr, nullptr, nullptr,
      nullptr, nullptr, lin_b, nullptr, (float*)d_out, NP);
}
